// SelfAttention_64433099375050
// MI455X (gfx1250) — matrix operand order repaired and checked by reading, not compiled, not run on hardware
//
#include <hip/hip_runtime.h>

typedef __attribute__((ext_vector_type(16))) __bf16 bf16x16;
typedef __attribute__((ext_vector_type(8)))  float  v8f;
typedef unsigned int u32x4 __attribute__((ext_vector_type(4)));
typedef int          i32x8 __attribute__((ext_vector_type(8)));
typedef int          i32x4 __attribute__((ext_vector_type(4)));

#define E_DIM 1024
#define H_DIM 16
#define D_DIM 64
#define L_DIM 512
#define N_DIM 8

#if __has_builtin(__builtin_amdgcn_tensor_load_to_lds) && \
    __has_builtin(__builtin_amdgcn_s_wait_tensorcnt)
#define HAVE_TDM 1
#else
#define HAVE_TDM 0
#endif

// ---------------------------------------------------------------------------
// WMMA helpers (CDNA5, wave32).  D = A(16x32 bf16) * B(32x16 bf16) + C(f32)
// ---------------------------------------------------------------------------
__device__ __forceinline__ v8f wmma_bf16(bf16x16 a, bf16x16 b, v8f c) {
  return __builtin_amdgcn_wmma_f32_16x16x32_bf16(false, a, false, b, (short)0, c,
                                                 false, false);
}

// A fragment from row-major [m][k] bf16 (16x32 tile at m0,k0).
__device__ __forceinline__ bf16x16 frag_a_bf16(const __bf16* base, int stride,
                                               int m0, int k0, int lane) {
  const __bf16* p = base + (size_t)(m0 + (lane & 15)) * stride + k0 + ((lane >> 4) << 4);
  bf16x16 a;
#pragma unroll
  for (int j = 0; j < 8; ++j) a[j] = p[j];
#pragma unroll
  for (int j = 0; j < 8; ++j) a[8 + j] = p[8 + j];
  return a;
}

// A fragment from row-major f32 with on-the-fly scale + bf16 convert.
__device__ __forceinline__ bf16x16 frag_a_f32(const float* base, int stride,
                                              int m0, int k0, int lane, float scale) {
  const float* p = base + (size_t)(m0 + (lane & 15)) * stride + k0 + ((lane >> 4) << 4);
  bf16x16 a;
#pragma unroll
  for (int j = 0; j < 8; ++j) a[j] = (__bf16)(p[j] * scale);
#pragma unroll
  for (int j = 0; j < 8; ++j) a[8 + j] = (__bf16)(p[8 + j] * scale);
  return a;
}

// B fragment from row-major B [k][n] (32x16 tile at k0,n0).
__device__ __forceinline__ bf16x16 frag_b_bf16(const __bf16* base, int stride,
                                               int k0, int n0, int lane) {
  const __bf16* p = base + (size_t)(k0 + ((lane >> 4) << 4)) * stride + n0 + (lane & 15);
  bf16x16 b;
#pragma unroll
  for (int j = 0; j < 16; ++j) b[j] = p[(size_t)j * stride];
  return b;
}

// B fragment where memory holds B^T row-major, i.e. [n][k] (contiguous loads).
__device__ __forceinline__ bf16x16 frag_bT_bf16(const __bf16* base, int stride,
                                                int k0, int n0, int lane) {
  const __bf16* p = base + (size_t)(n0 + (lane & 15)) * stride + k0 + ((lane >> 4) << 4);
  bf16x16 b;
#pragma unroll
  for (int j = 0; j < 16; ++j) b[j] = p[j];
  return b;
}

// ---------------------------------------------------------------------------
// Tensor Data Mover: DMA a [rows x 64] bf16 tile (row stride 1024 elem in
// memory) into LDS with padded row stride 72 bf16 (32 DW data + 4 DW pad).
// D# per CDNA5 ISA.  Issue from ONE wave only.
// This toolchain's builtin takes 6 args (g0,g1,g2,g3,extra,cpol).
// ---------------------------------------------------------------------------
#if HAVE_TDM
__device__ __forceinline__ void tdm_load_bf16_tile(unsigned lds_off,
                                                   const __bf16* gsrc, int rows) {
  unsigned long long ga = (unsigned long long)gsrc;
  u32x4 g0;
  g0[0] = 1u;                                   // count=1, user descriptor
  g0[1] = lds_off;                              // lds_addr (bytes)
  g0[2] = (unsigned)ga;                         // global_addr[31:0]
  g0[3] = (unsigned)((ga >> 32) & 0x01FFFFFFull) | 0x80000000u;  // [56:32]|type=2
  i32x8 g1;
  // data_size=1(2B) | pad_enable | pad_interval=4(32 DW) | pad_amount=3(4 DW)
  g1[0] = 0x07110000;
  g1[1] = (int)0xFFFF0000u;                     // tensor_dim0 = 0xFFFFFFFF (lo16)
  g1[2] = (int)0xFFFFFFFFu;                     // tensor_dim0 hi16 | tensor_dim1 lo16
  g1[3] = (int)((64u << 16) | 0xFFFFu);         // tensor_dim1 hi16 | tile_dim0=64
  g1[4] = rows;                                 // tile_dim1=rows, tile_dim2=0
  g1[5] = E_DIM;                                // tensor_dim0_stride = 1024 elements
  g1[6] = 0;
  g1[7] = 0;
  i32x4 z4 = (i32x4){0, 0, 0, 0};
  i32x8 z8 = (i32x8){0, 0, 0, 0, 0, 0, 0, 0};
  __builtin_amdgcn_tensor_load_to_lds(g0, g1, z4, z4, z8, 0);
}
#endif

// ---------------------------------------------------------------------------
// f32 -> bf16 elementwise convert
// ---------------------------------------------------------------------------
__global__ __launch_bounds__(256) void cvt_bf16_kernel(const float* __restrict__ in,
                                                       __bf16* __restrict__ out, int n) {
  int i = blockIdx.x * blockDim.x + threadIdx.x;
  if (i < n) out[i] = (__bf16)in[i];
}

// ---------------------------------------------------------------------------
// GEMM:  C[M,N] = A[M,K] * B[K,N] + bias[N]
// 128x64 tile / block, BK=32, 256 threads (8 waves in 4x2), each wave a 32x32
// sub-tile: 2 A-frags x 2 B-frags -> 4 WMMAs with operand reuse.
// ---------------------------------------------------------------------------
#define GBM 128
#define GBN 64
#define GBK 32

template <bool A_BF16, bool OUT_F32>
__global__ __launch_bounds__(256) void gemm_bias_kernel(
    const void* __restrict__ Ap, const float* __restrict__ Bw,
    const float* __restrict__ bias, void* __restrict__ Cp, int M, int Nn, int K) {
  __shared__ __bf16 As[GBM][GBK + 8];   // stride 40 (bank-spread)
  __shared__ __bf16 Bs[GBK][GBN + 4];   // stride 68

  const int tid  = threadIdx.x;
  const int lane = tid & 31;
  const int wid  = tid >> 5;
  const int wm   = (wid & 3) << 5;      // wave row offset (0,32,64,96)
  const int wn   = (wid >> 2) << 5;     // wave col offset (0,32)
  const int m_blk = blockIdx.y * GBM;
  const int n_blk = blockIdx.x * GBN;

  const float*  Af = (const float*)Ap;
  const __bf16* Ab = (const __bf16*)Ap;

  v8f acc[2][2];
#pragma unroll
  for (int i = 0; i < 2; ++i)
#pragma unroll
    for (int j = 0; j < 2; ++j)
      acc[i][j] = (v8f){0.f, 0.f, 0.f, 0.f, 0.f, 0.f, 0.f, 0.f};

  for (int k0 = 0; k0 < K; k0 += GBK) {
    __syncthreads();
    for (int i = tid; i < GBM * GBK; i += 256) {
      int r = i >> 5, c = i & 31;
      float v = A_BF16 ? (float)Ab[(size_t)(m_blk + r) * K + k0 + c]
                       : Af[(size_t)(m_blk + r) * K + k0 + c];
      As[r][c] = (__bf16)v;
    }
    for (int i = tid; i < GBK * GBN; i += 256) {
      int r = i >> 6, c = i & 63;
      Bs[r][c] = (__bf16)Bw[(size_t)(k0 + r) * Nn + n_blk + c];
    }
    // prefetch next k-tile while this one computes (global_prefetch_b8)
    if (k0 + GBK < K) {
      if (tid < GBM) {
        size_t idx = (size_t)(m_blk + tid) * K + k0 + GBK;
        const char* pa = A_BF16 ? (const char*)(Ab + idx) : (const char*)(Af + idx);
        __builtin_prefetch(pa, 0, 1);
      }
      if (tid < GBK)
        __builtin_prefetch(&Bw[(size_t)(k0 + GBK + tid) * Nn + n_blk], 0, 1);
    }
    __syncthreads();
    bf16x16 a0 = frag_a_bf16(&As[0][0], GBK + 8, wm, 0, lane);
    bf16x16 a1 = frag_a_bf16(&As[0][0], GBK + 8, wm + 16, 0, lane);
    bf16x16 b0 = frag_b_bf16(&Bs[0][0], GBN + 4, 0, wn, lane);
    bf16x16 b1 = frag_b_bf16(&Bs[0][0], GBN + 4, 0, wn + 16, lane);
    acc[0][0] = wmma_bf16(a0, b0, acc[0][0]);
    acc[0][1] = wmma_bf16(a0, b1, acc[0][1]);
    acc[1][0] = wmma_bf16(a1, b0, acc[1][0]);
    acc[1][1] = wmma_bf16(a1, b1, acc[1][1]);
  }

  const int mrow = (lane >> 4) << 3;
  const int ncol = lane & 15;
#pragma unroll
  for (int i = 0; i < 2; ++i) {
#pragma unroll
    for (int j = 0; j < 2; ++j) {
      int gn = n_blk + wn + j * 16 + ncol;
      float bia = bias[gn];
#pragma unroll
      for (int r = 0; r < 8; ++r) {
        int gm = m_blk + wm + i * 16 + mrow + r;
        float v = acc[i][j][r] + bia;
        if (OUT_F32) ((float*)Cp)[(size_t)gm * Nn + gn] = v;
        else         ((__bf16*)Cp)[(size_t)gm * Nn + gn] = (__bf16)v;
      }
    }
  }
}

// ---------------------------------------------------------------------------
// Attention: one block per (n, h, 16-query tile); 256 threads = 8 waves.
//   S[16,512] = Q K^T + skewed rel bias ; softmax ; O = P V ; *after_mask
// Rel trick: T[qi,p] = sum_d Q[qi,d]*rel_emb[pmin+p, hD+d]  (WMMA GEMM),
//            S[qi,k] += T[qi, k - qi + 15], pmin = 512 + kb0 - qbase - 15
// K/V/Q tiles are DMA'd into LDS by the Tensor Data Mover (wave 0 issues,
// waits TENSORcnt, workgroup barrier publishes).
// ---------------------------------------------------------------------------
__global__ __launch_bounds__(256) void attn_kernel(
    const __bf16* __restrict__ qb, const __bf16* __restrict__ kbp,
    const __bf16* __restrict__ vbp, const __bf16* __restrict__ relb,
    const int* __restrict__ mask, const float* __restrict__ after_mask,
    __bf16* __restrict__ ctx) {
  __shared__ __bf16 Qs[16][72];
  __shared__ __bf16 Ks[64][72];
  __shared__ __bf16 Vs[64][72];
  __shared__ float  Sc[16][512];
  __shared__ float  Tb[16][84];
  __shared__ float  pred[16][16];
  __shared__ float  rowmax_s[16];
  __shared__ float  rinv_s[16];

  const int tid   = threadIdx.x;
  const int lane  = tid & 31;
  const int wid   = tid >> 5;
  const int qbase = blockIdx.x * 16;
  const int h     = blockIdx.y;
  const int n     = blockIdx.z;

  const __bf16* qp = qb  + (size_t)n * L_DIM * E_DIM + (size_t)h * D_DIM;
  const __bf16* kp = kbp + (size_t)n * L_DIM * E_DIM + (size_t)h * D_DIM;
  const __bf16* vp = vbp + (size_t)n * L_DIM * E_DIM + (size_t)h * D_DIM;

#if HAVE_TDM
  const unsigned qs_off = (unsigned)(unsigned long long)(void*)&Qs[0][0];
  const unsigned ks_off = (unsigned)(unsigned long long)(void*)&Ks[0][0];
  const unsigned vs_off = (unsigned)(unsigned long long)(void*)&Vs[0][0];
  if (wid == 0) tdm_load_bf16_tile(qs_off, qp + (size_t)qbase * E_DIM, 16);
#else
  for (int i = tid; i < 16 * D_DIM; i += 256) {
    int r = i >> 6, c = i & 63;
    Qs[r][c] = qp[(size_t)(qbase + r) * E_DIM + c];
  }
#endif

  // ---- scores: loop over key blocks of 64 ----
  for (int kb0 = 0; kb0 < L_DIM; kb0 += 64) {
    __syncthreads();  // previous readers of Ks done
#if HAVE_TDM
    if (wid == 0) {
      tdm_load_bf16_tile(ks_off, kp + (size_t)kb0 * E_DIM, 64);
      __builtin_amdgcn_s_wait_tensorcnt(0);   // covers Qs DMA too (in-order)
    }
#else
    for (int i = tid; i < 64 * D_DIM; i += 256) {
      int r = i >> 6, c = i & 63;
      Ks[r][c] = kp[(size_t)(kb0 + r) * E_DIM + c];
    }
#endif
    __syncthreads();
    const int pmin = 512 + kb0 - qbase - 15;
    // 9 tile jobs: 0..3 = QK^T 16x16 tiles, 4..8 = rel-T 16x16 tiles
    for (int job = wid; job < 9; job += 8) {
      v8f acc = (v8f){0.f, 0.f, 0.f, 0.f, 0.f, 0.f, 0.f, 0.f};
      bf16x16 a0 = frag_a_bf16(&Qs[0][0], 72, 0, 0, lane);
      bf16x16 a1 = frag_a_bf16(&Qs[0][0], 72, 0, 32, lane);
      int mrow = (lane >> 4) << 3;
      int ncol = lane & 15;
      if (job < 4) {
        int nt = job * 16;
        bf16x16 b0 = frag_bT_bf16(&Ks[0][0], 72, 0, nt, lane);
        bf16x16 b1 = frag_bT_bf16(&Ks[0][0], 72, 32, nt, lane);
        acc = wmma_bf16(a0, b0, acc);
        acc = wmma_bf16(a1, b1, acc);
#pragma unroll
        for (int r = 0; r < 8; ++r) Sc[mrow + r][kb0 + nt + ncol] = acc[r];
      } else {
        int pt = (job - 4) * 16;
        int prow = pmin + pt + ncol;
        prow = prow < 0 ? 0 : (prow > 1023 ? 1023 : prow);  // pad cols only
        const __bf16* bp = relb + (size_t)prow * E_DIM + h * D_DIM + ((lane >> 4) << 4);
        bf16x16 b0, b1;
#pragma unroll
        for (int j = 0; j < 16; ++j) b0[j] = bp[j];
#pragma unroll
        for (int j = 0; j < 16; ++j) b1[j] = bp[32 + j];
        acc = wmma_bf16(a0, b0, acc);
        acc = wmma_bf16(a1, b1, acc);
#pragma unroll
        for (int r = 0; r < 8; ++r) Tb[mrow + r][pt + ncol] = acc[r];
      }
    }
    __syncthreads();
    // skew gather-add:  S[qi, kb0+kj] += T[qi, kj - qi + 15]
    for (int i = tid; i < 16 * 64; i += 256) {
      int qi = i >> 6, kj = i & 63;
      Sc[qi][kb0 + kj] += Tb[qi][kj - qi + 15];
    }
  }
  __syncthreads();

  // ---- softmax over 512 keys, mask + 1/sqrt(64) scale (16 threads/row) ----
  {
    const int row = tid >> 4;
    const int sub = tid & 15;
    const int* mrow = mask + ((size_t)n * L_DIM + (qbase + row)) * L_DIM;
    float mx = -3.0e38f;
    for (int c = sub * 32; c < sub * 32 + 32; ++c) {
      float e = (mrow[c] != 0) ? Sc[row][c] * 0.125f : -1.25e19f;
      Sc[row][c] = e;
      mx = fmaxf(mx, e);
    }
    pred[row][sub] = mx;
    __syncthreads();
    if (tid < 16) {
      float m2 = pred[tid][0];
#pragma unroll
      for (int j = 1; j < 16; ++j) m2 = fmaxf(m2, pred[tid][j]);
      rowmax_s[tid] = m2;
    }
    __syncthreads();
    float rmax = rowmax_s[row];
    float sum = 0.f;
    for (int c = sub * 32; c < sub * 32 + 32; ++c) {
      float e = __expf(Sc[row][c] - rmax);
      Sc[row][c] = e;
      sum += e;
    }
    pred[row][sub] = sum;
    __syncthreads();
    if (tid < 16) {
      float s2 = 0.f;
#pragma unroll
      for (int j = 0; j < 16; ++j) s2 += pred[tid][j];
      rinv_s[tid] = 1.0f / s2;
    }
  }

  // ---- O = P @ V (waves 0..3 own one 16x16 d-tile; 1/rowsum folded in) ----
  v8f oacc = (v8f){0.f, 0.f, 0.f, 0.f, 0.f, 0.f, 0.f, 0.f};
  for (int kb0 = 0; kb0 < L_DIM; kb0 += 64) {
    __syncthreads();
#if HAVE_TDM
    if (wid == 0) {
      tdm_load_bf16_tile(vs_off, vp + (size_t)kb0 * E_DIM, 64);
      __builtin_amdgcn_s_wait_tensorcnt(0);
    }
#else
    for (int i = tid; i < 64 * D_DIM; i += 256) {
      int r = i >> 6, c = i & 63;
      Vs[r][c] = vp[(size_t)(kb0 + r) * E_DIM + c];
    }
#endif
    __syncthreads();
    if (wid < 4) {
      float rscale = rinv_s[lane & 15];
      bf16x16 a0 = frag_a_f32(&Sc[0][0], 512, 0, kb0, lane, rscale);
      bf16x16 a1 = frag_a_f32(&Sc[0][0], 512, 0, kb0 + 32, lane, rscale);
      bf16x16 b0 = frag_b_bf16(&Vs[0][0], 72, 0, wid * 16, lane);
      bf16x16 b1 = frag_b_bf16(&Vs[0][0], 72, 32, wid * 16, lane);
      oacc = wmma_bf16(a0, b0, oacc);
      oacc = wmma_bf16(a1, b1, oacc);
    }
  }
  if (wid < 4) {
    int mrow = (lane >> 4) << 3;
    int ncol = lane & 15;
#pragma unroll
    for (int r = 0; r < 8; ++r) {
      int q = qbase + mrow + r;
      float am = after_mask[(size_t)n * L_DIM + q];
      ctx[((size_t)n * L_DIM + q) * E_DIM + h * D_DIM + wid * 16 + ncol] =
          (__bf16)(oacc[r] * am);
    }
  }
}

// ---------------------------------------------------------------------------
// launch
// ---------------------------------------------------------------------------
extern "C" void kernel_launch(void* const* d_in, const int* in_sizes, int n_in,
                              void* d_out, int out_size, void* d_ws, size_t ws_size,
                              hipStream_t stream) {
  const float* values    = (const float*)d_in[0];
  const float* keys      = (const float*)d_in[1];
  const float* queries   = (const float*)d_in[2];
  const int*   mask      = (const int*)d_in[3];
  const float* after_m   = (const float*)d_in[4];
  const float* Wv        = (const float*)d_in[5];
  const float* bv        = (const float*)d_in[6];
  const float* Wk        = (const float*)d_in[7];
  const float* bk        = (const float*)d_in[8];
  const float* Wq        = (const float*)d_in[9];
  const float* bq        = (const float*)d_in[10];
  const float* Wo        = (const float*)d_in[11];
  const float* bo        = (const float*)d_in[12];
  const float* rel_emb   = (const float*)d_in[13];
  float* out = (float*)d_out;

  const int M = N_DIM * L_DIM;  // 4096
  const size_t sz_qkv = (size_t)M * E_DIM * sizeof(__bf16);          // 8 MB each
  const size_t sz_rel = (size_t)2 * L_DIM * E_DIM * sizeof(__bf16);  // 2 MB
  if (ws_size < 4 * sz_qkv + sz_rel) return;

  char* ws = (char*)d_ws;
  __bf16* qb   = (__bf16*)(ws);
  __bf16* kb_  = (__bf16*)(ws + sz_qkv);
  __bf16* vb_  = (__bf16*)(ws + 2 * sz_qkv);
  __bf16* ctx  = (__bf16*)(ws + 3 * sz_qkv);
  __bf16* relb = (__bf16*)(ws + 4 * sz_qkv);

  {
    int n = 2 * L_DIM * E_DIM;
    cvt_bf16_kernel<<<(n + 255) / 256, 256, 0, stream>>>(rel_emb, relb, n);
  }

  dim3 ggrid(E_DIM / GBN, M / GBM);  // (16, 32)
  gemm_bias_kernel<false, false><<<ggrid, 256, 0, stream>>>(queries, Wq, bq, qb,  M, E_DIM, E_DIM);
  gemm_bias_kernel<false, false><<<ggrid, 256, 0, stream>>>(keys,    Wk, bk, kb_, M, E_DIM, E_DIM);
  gemm_bias_kernel<false, false><<<ggrid, 256, 0, stream>>>(values,  Wv, bv, vb_, M, E_DIM, E_DIM);

  attn_kernel<<<dim3(L_DIM / 16, H_DIM, N_DIM), 256, 0, stream>>>(
      qb, kb_, vb_, relb, mask, after_m, ctx);

  gemm_bias_kernel<true, true><<<ggrid, 256, 0, stream>>>(ctx, Wo, bo, out, M, E_DIM, E_DIM);
}
